// RPN_48919677501804
// MI455X (gfx1250) — matrix operand order repaired and checked by reading, not compiled, not run on hardware
//
#include <hip/hip_runtime.h>
#include <hip/hip_bf16.h>
#include <stdint.h>

#define H_FEAT 84
#define W_FEAT 125
#define HW     10500          // 84*125
#define CIN    512
#define COUT   512
#define NA     9
#define NANCH  (HW*NA)        // 94500
#define PRE    6000
#define POST   300
#define CAP    8192           // pow2 >= PRE (bitonic sort size)
#define NMSB   ((PRE+63)/64)  // 94 mask words per row

#if defined(__has_builtin)
#if __has_builtin(__builtin_amdgcn_global_load_async_to_lds_b128) && \
    __has_builtin(__builtin_amdgcn_s_wait_asynccnt)
#define HAVE_ASYNC_LDS 1
#endif
#endif

#define AS1 __attribute__((address_space(1)))
#define AS3 __attribute__((address_space(3)))

typedef int v4i __attribute__((vector_size(16)));  // matches builtin's int4 param type

typedef __attribute__((ext_vector_type(16))) _Float16 v16h;
typedef __attribute__((ext_vector_type(8)))  float    v8f;

union AF { v16h v; unsigned u[8]; };
union CF { v8f  v; float    f[8]; };

// ---------------- feature transpose: [ci][pix] f32 -> [pix][ci] f16 ----------------
__global__ __launch_bounds__(1024) void k_tr_feat(const float* __restrict__ in,
                                                  _Float16* __restrict__ out) {
  __shared__ _Float16 tile[32][33];
  const int p0 = blockIdx.x * 32;  // pixel base
  const int c0 = blockIdx.y * 32;  // channel base
  const int tx = threadIdx.x & 31;
  const int ty = threadIdx.x >> 5;
  const int p = p0 + tx;
  tile[ty][tx] = (p < HW) ? (_Float16)in[(size_t)(c0 + ty) * HW + p] : (_Float16)0.0f;
  __syncthreads();
  const int pp = p0 + ty;
  if (pp < HW) out[(size_t)pp * CIN + c0 + tx] = tile[tx][ty];
}

// repack W1 (OIHW 512,512,3,3) -> 9 tap matrices w1h[t][co][ci], f16
__global__ void k_cvt_w1(const float* __restrict__ in, _Float16* __restrict__ out) {
  int i = blockIdx.x * 256 + threadIdx.x;
  if (i < 9 * COUT * CIN) {
    int t   = i / (COUT * CIN);
    int rem = i % (COUT * CIN);
    int co  = rem / CIN;
    int ci  = rem % CIN;
    out[i] = (_Float16)in[(co * CIN + ci) * 9 + t];
  }
}

// zero 64-byte pad used as the async-copy source for out-of-image im2col columns
__global__ void k_zeropad(unsigned* __restrict__ zp) {
  if (threadIdx.x < 16) zp[threadIdx.x] = 0u;
}

// ---------------- conv3x3 + relu as implicit GEMM (WMMA f16->f32) ----------------
// D[co][pix] = relu( sum_tap sum_ci W[t][co][ci] * Ftr[shift_t(pix)][ci] + b1[co] )
// block tile 128(M=cout) x 128(N=pix); 8 waves of 64x32; K-step 32.
__global__ __launch_bounds__(256) void k_conv(const _Float16* __restrict__ ftr,
                                              const _Float16* __restrict__ w1h,
                                              const float* __restrict__ b1,
                                              const _Float16* __restrict__ zeropad,
                                              _Float16* __restrict__ ct) {
  __shared__ __align__(16) _Float16 As[128 * 32];  // [row m][k]
  __shared__ __align__(16) _Float16 Bs[128 * 32];  // [col n][k]  (k-fastest for pair reads)

  const int tid   = threadIdx.x;
  const int lane  = tid & 31;
  const int wid   = tid >> 5;
  const int m0    = (wid >> 2) * 64;   // wave M offset in tile
  const int n0    = (wid & 3) * 32;    // wave N offset in tile
  const int blockN = blockIdx.x * 128; // pixel base
  const int blockM = blockIdx.y * 128; // cout base

  CF acc[4][2];
#pragma unroll
  for (int a = 0; a < 4; ++a)
#pragma unroll
    for (int b = 0; b < 2; ++b)
#pragma unroll
      for (int q = 0; q < 8; ++q) acc[a][b].f[q] = 0.0f;

  // this thread's staging assignment (same pattern for A and B tiles)
  const int nn   = tid >> 1;            // 0..127 tile row/col
  const int kh   = (tid & 1) * 16;      // 0 or 16 (16 halfs = 2x uint4)
  const int npix = blockN + nn;
  const int py   = npix / W_FEAT;
  const int px   = npix % W_FEAT;

#if !defined(HAVE_ASYNC_LDS)
  const uint4 zero4 = make_uint4(0u, 0u, 0u, 0u);
#endif
  _Float16* const aDst = &As[nn * 32 + kh];
  _Float16* const bDst = &Bs[nn * 32 + kh];

  for (int tap = 0; tap < 9; ++tap) {
    const int r = tap / 3, s = tap % 3;
    const int yy = py + r - 1;
    const int xx = px + s - 1;
    const bool pvalid = (npix < HW) && (yy >= 0) && (yy < H_FEAT) && (xx >= 0) && (xx < W_FEAT);
    const int spix = pvalid ? (yy * W_FEAT + xx) : 0;
    const _Float16* wt    = w1h + (size_t)tap * (COUT * CIN);
    const _Float16* bcol  = ftr + (size_t)spix * CIN;  // + kb + kh per step

    for (int kb = 0; kb < CIN; kb += 32) {
      const _Float16* asrc = wt + (size_t)(blockM + nn) * CIN + kb + kh;
      const _Float16* bsrc = bcol + kb + kh;
#if defined(HAVE_ASYNC_LDS)
      // CDNA5 async DMA: global -> LDS, tracked with ASYNCcnt (no VGPR data)
      const _Float16* bs_src = pvalid ? bsrc : zeropad;
      __builtin_amdgcn_global_load_async_to_lds_b128(
          (AS1 v4i*)(uintptr_t)asrc, (AS3 v4i*)(uintptr_t)aDst, 0, 0);
      __builtin_amdgcn_global_load_async_to_lds_b128(
          (AS1 v4i*)(uintptr_t)(asrc + 8), (AS3 v4i*)(uintptr_t)(aDst + 8), 0, 0);
      __builtin_amdgcn_global_load_async_to_lds_b128(
          (AS1 v4i*)(uintptr_t)bs_src, (AS3 v4i*)(uintptr_t)bDst, 0, 0);
      __builtin_amdgcn_global_load_async_to_lds_b128(
          (AS1 v4i*)(uintptr_t)(bs_src + 8), (AS3 v4i*)(uintptr_t)(bDst + 8), 0, 0);
      if (kb + 32 < CIN) {
        __builtin_prefetch(asrc + 32, 0, 3);
        __builtin_prefetch(bsrc + 32, 0, 3);
      }
      __builtin_amdgcn_s_wait_asynccnt(0);
#else
      // register staging: unconditional global_load_b128, zero-select on data
      const uint4 a0 = *(const uint4*)(asrc + 0);
      const uint4 a1 = *(const uint4*)(asrc + 8);
      uint4 b0v = *(const uint4*)(bsrc + 0);
      uint4 b1v = *(const uint4*)(bsrc + 8);
      if (!pvalid) { b0v = zero4; b1v = zero4; }
      *(uint4*)(aDst + 0) = a0;
      *(uint4*)(aDst + 8) = a1;
      *(uint4*)(bDst + 0) = b0v;
      *(uint4*)(bDst + 8) = b1v;
      if (kb + 32 < CIN) {
        __builtin_prefetch(asrc + 32, 0, 3);
        __builtin_prefetch(bsrc + 32, 0, 3);
      }
#endif
      __syncthreads();

      // A fragments: lane = M row (0-15), upper half-wave takes K+8 interleave
      AF afr[4];
      const int hh = lane >> 4;
      const int rl = lane & 15;
#pragma unroll
      for (int mi = 0; mi < 4; ++mi) {
        const int row = m0 + mi * 16 + rl;
#pragma unroll
        for (int v = 0; v < 8; ++v) {
          const int kb2 = (v < 4 ? 2 * v : 16 + 2 * (v - 4)) + 8 * hh;
          afr[mi].u[v] = *(const unsigned*)&As[row * 32 + kb2];
        }
      }
      // B fragments: lane = N col, upper half-wave takes K+8 interleave
      AF bfr[2];
#pragma unroll
      for (int ni = 0; ni < 2; ++ni) {
        const int col = n0 + ni * 16 + rl;
#pragma unroll
        for (int v = 0; v < 8; ++v) {
          const int kb2 = (v < 4 ? 2 * v : 16 + 2 * (v - 4)) + 8 * hh;
          bfr[ni].u[v] = *(const unsigned*)&Bs[col * 32 + kb2];
        }
      }
#pragma unroll
      for (int mi = 0; mi < 4; ++mi)
#pragma unroll
        for (int ni = 0; ni < 2; ++ni)
          acc[mi][ni].v = __builtin_amdgcn_wmma_f32_16x16x32_f16(
              false, afr[mi].v, false, bfr[ni].v, (short)0, acc[mi][ni].v, false, false);
      __syncthreads();
    }
  }

  // epilogue: bias + relu + f16, stored transposed ct[pix][co]
  const int hh = lane >> 4;
  const int cl = lane & 15;
#pragma unroll
  for (int mi = 0; mi < 4; ++mi)
#pragma unroll
    for (int ni = 0; ni < 2; ++ni) {
      const int col = blockN + n0 + ni * 16 + cl;
      if (col < HW) {
#pragma unroll
        for (int i = 0; i < 8; ++i) {
          const int row = blockM + m0 + mi * 16 + i + 8 * hh;
          float v = acc[mi][ni].f[i] + b1[row];
          v = v > 0.0f ? v : 0.0f;
          ct[(size_t)col * COUT + row] = (_Float16)v;
        }
      }
    }
}

// ---------------- 1x1 convs + softmax + anchor decode + filter ----------------
__global__ __launch_bounds__(64) void k_props(const _Float16* __restrict__ ct,
                                              const float* __restrict__ Ws, const float* __restrict__ bs,
                                              const float* __restrict__ Wb, const float* __restrict__ bb,
                                              const float* __restrict__ im_info,
                                              float* __restrict__ scores, float* __restrict__ boxes) {
  __shared__ __align__(16) _Float16 col[512];
  __shared__ float outv[64];
  const int n = blockIdx.x;
  const int t = threadIdx.x;
  ((uint4*)col)[t] = ((const uint4*)(ct + (size_t)n * 512))[t];  // 64*16B = 1KB column
  __syncthreads();
  if (t < 54) {
    const float* w = (t < 18) ? (Ws + (size_t)t * 512) : (Wb + (size_t)(t - 18) * 512);
    float acc = (t < 18) ? bs[t] : bb[t - 18];
#pragma unroll 4
    for (int c = 0; c < 512; ++c) acc += (float)col[c] * w[c];
    outv[t] = acc;
  }
  __syncthreads();
  if (t < 9) {
    const int a = t;
    const float s0 = outv[2 * a], s1 = outv[2 * a + 1];
    const float fg = 1.0f / (1.0f + __expf(s0 - s1));
    const float dx = outv[18 + 4 * a + 0];
    const float dy = outv[18 + 4 * a + 1];
    float dw = outv[18 + 4 * a + 2];
    float dh = outv[18 + 4 * a + 3];
    dw = fminf(fmaxf(dw, -10.0f), 10.0f);
    dh = fminf(fmaxf(dh, -10.0f), 10.0f);
    const float AW[9] = {184.f, 368.f, 736.f, 128.f, 256.f, 512.f, 88.f, 176.f, 352.f};
    const float AH[9] = {96.f, 192.f, 384.f, 128.f, 256.f, 512.f, 176.f, 352.f, 704.f};
    const int pyy = n / W_FEAT, pxx = n % W_FEAT;
    const float acx = 8.0f + pxx * 16.0f;
    const float acy = 8.0f + pyy * 16.0f;
    const float aw = AW[a], ah = AH[a];
    const float pcx = dx * aw + acx, pcy = dy * ah + acy;
    const float pw = __expf(dw) * aw, ph = __expf(dh) * ah;
    const float imh = im_info[0], imw = im_info[1], sc = im_info[2];
    const float x1 = fminf(fmaxf(pcx - 0.5f * pw, 0.0f), imw - 1.0f);
    const float y1 = fminf(fmaxf(pcy - 0.5f * ph, 0.0f), imh - 1.0f);
    const float x2 = fminf(fmaxf(pcx + 0.5f * pw, 0.0f), imw - 1.0f);
    const float y2 = fminf(fmaxf(pcy + 0.5f * ph, 0.0f), imh - 1.0f);
    const float msz = 16.0f * sc;
    const bool ok = (x2 - x1 + 1.0f >= msz) && (y2 - y1 + 1.0f >= msz);
    const int i = n * 9 + a;
    scores[i] = ok ? fg : -1.0f;
    boxes[4 * i + 0] = x1;
    boxes[4 * i + 1] = y1;
    boxes[4 * i + 2] = x2;
    boxes[4 * i + 3] = y2;
  }
}

__device__ __forceinline__ unsigned fkey(float f) {
  unsigned u = __float_as_uint(f);
  return u ^ ((u >> 31) ? 0xFFFFFFFFu : 0x80000000u);  // monotone: bigger float -> bigger key
}

// ---------------- radix-select threshold for top-PRE ----------------
__global__ __launch_bounds__(1024) void k_select(const float* __restrict__ scores, unsigned* __restrict__ scal) {
  __shared__ unsigned hist[256];
  __shared__ unsigned sh_prefix;
  __shared__ int sh_need;
  const int t = threadIdx.x;
  if (t == 0) { sh_prefix = 0; sh_need = PRE; }
  __syncthreads();
  for (int level = 3; level >= 0; --level) {
    for (int d = t; d < 256; d += 1024) hist[d] = 0;
    __syncthreads();
    const unsigned prefix = sh_prefix;
    const unsigned hmask = (level == 3) ? 0u : (0xFFFFFFFFu << ((level + 1) * 8));
    for (int i = t; i < NANCH; i += 1024) {
      const unsigned k = fkey(scores[i]);
      if (((k ^ prefix) & hmask) == 0) atomicAdd(&hist[(k >> (level * 8)) & 255], 1u);
    }
    __syncthreads();
    if (t == 0) {
      const int need = sh_need;
      unsigned cum = 0;
      int d;
      for (d = 255; d >= 0; --d) {
        if (cum + hist[d] >= (unsigned)need) break;
        cum += hist[d];
      }
      if (d < 0) d = 0;
      sh_prefix = prefix | ((unsigned)d << (level * 8));
      sh_need = need - (int)cum;
    }
    __syncthreads();
  }
  if (t == 0) { scal[0] = sh_prefix; scal[1] = (unsigned)sh_need; scal[2] = 0; }
}

__global__ void k_compact(const float* __restrict__ scores, const unsigned* __restrict__ scal,
                          unsigned long long* __restrict__ pairs, unsigned* __restrict__ cnt) {
  const unsigned T = scal[0];
  for (int i = blockIdx.x * blockDim.x + threadIdx.x; i < NANCH; i += gridDim.x * blockDim.x) {
    const unsigned k = fkey(scores[i]);
    if (k >= T) {
      const unsigned p = atomicAdd(cnt, 1u);
      if (p < CAP) pairs[p] = (((unsigned long long)(~k)) << 32) | (unsigned)i;  // (desc score, asc idx)
    }
  }
}

// ---------------- bitonic sort of candidates + gather sorted proposals ----------------
__global__ __launch_bounds__(1024) void k_sortgather(const unsigned long long* __restrict__ pairs,
                                                     const unsigned* __restrict__ scal,
                                                     const float* __restrict__ boxes,
                                                     float* __restrict__ props) {
  __shared__ unsigned long long sk[CAP];  // 64 KB LDS (CDNA5: up to 320KB/WG)
  const int t = threadIdx.x;
  unsigned cnt = scal[2];
  if (cnt > CAP) cnt = CAP;
  for (int i = t; i < CAP; i += 1024)
    sk[i] = (i < (int)cnt) ? pairs[i] : 0xFFFFFFFFFFFFFFFFull;
  __syncthreads();
  for (int k = 2; k <= CAP; k <<= 1)
    for (int j = k >> 1; j > 0; j >>= 1) {
      for (int i = t; i < CAP; i += 1024) {
        const int ixj = i ^ j;
        if (ixj > i) {
          const unsigned long long a = sk[i], b = sk[ixj];
          const bool up = ((i & k) == 0);
          if ((a > b) == up) { sk[i] = b; sk[ixj] = a; }
        }
      }
      __syncthreads();
    }
  for (int jj = t; jj < PRE; jj += 1024) {
    unsigned idx = (unsigned)(sk[jj] & 0xFFFFFFFFu);
    if (idx >= NANCH) idx = 0;
    props[jj * 4 + 0] = boxes[idx * 4 + 0];
    props[jj * 4 + 1] = boxes[idx * 4 + 1];
    props[jj * 4 + 2] = boxes[idx * 4 + 2];
    props[jj * 4 + 3] = boxes[idx * 4 + 3];
  }
}

// ---------------- NMS suppression bitmask ----------------
__global__ __launch_bounds__(64) void k_nms(const float* __restrict__ props,
                                            unsigned long long* __restrict__ mask) {
  const int rb = blockIdx.y, cb = blockIdx.x;
  const int t = threadIdx.x;
  const int i = rb * 64 + t;
  if (cb < rb) {  // only need j > i; keep row fully initialized
    if (i < PRE) mask[(size_t)i * NMSB + cb] = 0ull;
    return;
  }
  __shared__ float4 cbox[64];
  const int cj = cb * 64 + t;
  if (cj < PRE) cbox[t] = ((const float4*)props)[cj];
  __syncthreads();
  if (i >= PRE) return;
  const float4 bi = ((const float4*)props)[i];
  const float areai = (bi.z - bi.x + 1.0f) * (bi.w - bi.y + 1.0f);
  unsigned long long bits = 0ull;
  const int start = (rb == cb) ? t + 1 : 0;
  int cmax = PRE - cb * 64;
  if (cmax > 64) cmax = 64;
  for (int c = start; c < cmax; ++c) {
    const float4 bj = cbox[c];
    const float xx1 = fmaxf(bi.x, bj.x), yy1 = fmaxf(bi.y, bj.y);
    const float xx2 = fminf(bi.z, bj.z), yy2 = fminf(bi.w, bj.w);
    const float w = fmaxf(0.0f, xx2 - xx1 + 1.0f), h = fmaxf(0.0f, yy2 - yy1 + 1.0f);
    const float inter = w * h;
    const float areaj = (bj.z - bj.x + 1.0f) * (bj.w - bj.y + 1.0f);
    const float iou = inter / (areai + areaj - inter);
    if (iou > 0.7f) bits |= (1ull << c);
  }
  mask[(size_t)i * NMSB + cb] = bits;
}

// ---------------- sequential keep scan + ROI output ----------------
__global__ __launch_bounds__(256) void k_final(const float* __restrict__ props,
                                               const unsigned long long* __restrict__ mask,
                                               float* __restrict__ out) {
  __shared__ unsigned long long remv[NMSB];
  __shared__ int sel[POST];
  __shared__ int scnt;
  const int t = threadIdx.x;
  for (int i = t; i < NMSB; i += 256) remv[i] = 0ull;
  if (t == 0) scnt = 0;
  __syncthreads();
  if (t == 0) {
    int c = 0;
    for (int i = 0; i < PRE && c < POST; ++i) {
      if (!((remv[i >> 6] >> (i & 63)) & 1ull)) {
        sel[c++] = i;
        const unsigned long long* mr = mask + (size_t)i * NMSB;
        for (int b = i >> 6; b < NMSB; ++b) remv[b] |= mr[b];
      }
    }
    scnt = c;
  }
  __syncthreads();
  const int c = scnt;
  for (int j = t; j < POST; j += 256) {
    float x1 = 0.f, y1 = 0.f, x2 = 0.f, y2 = 0.f;
    if (j < c) {
      const float4 b = ((const float4*)props)[sel[j]];
      x1 = b.x; y1 = b.y; x2 = b.z; y2 = b.w;
    }
    out[j * 5 + 0] = 0.0f;
    out[j * 5 + 1] = x1;
    out[j * 5 + 2] = y1;
    out[j * 5 + 3] = x2;
    out[j * 5 + 4] = y2;
  }
}

extern "C" void kernel_launch(void* const* d_in, const int* in_sizes, int n_in,
                              void* d_out, int out_size, void* d_ws, size_t ws_size,
                              hipStream_t stream) {
  const float* features = (const float*)d_in[0];
  const float* im_info  = (const float*)d_in[1];
  const float* W1 = (const float*)d_in[2];
  const float* b1 = (const float*)d_in[3];
  const float* Ws = (const float*)d_in[4];
  const float* bs = (const float*)d_in[5];
  const float* Wb = (const float*)d_in[6];
  const float* bb = (const float*)d_in[7];
  float* out = (float*)d_out;

  char* ws = (char*)d_ws;
  size_t off = 0;
  auto alloc = [&](size_t bytes) -> char* {
    char* p = ws + off;
    off = (off + bytes + 255) & ~(size_t)255;
    return p;
  };
  _Float16* ftr  = (_Float16*)alloc((size_t)HW * CIN * 2);      // transposed features [pix][ci]
  _Float16* w1h  = (_Float16*)alloc((size_t)9 * COUT * CIN * 2);
  _Float16* ct   = (_Float16*)alloc((size_t)HW * COUT * 2);     // conv out [pix][co]
  float* scores  = (float*)alloc((size_t)NANCH * 4);
  float* boxes   = (float*)alloc((size_t)NANCH * 16);
  unsigned long long* pairs = (unsigned long long*)alloc((size_t)CAP * 8);
  float* props   = (float*)alloc((size_t)PRE * 16);
  unsigned long long* mask  = (unsigned long long*)alloc((size_t)PRE * NMSB * 8);
  unsigned* scal = (unsigned*)alloc(256);
  unsigned* zpad = (unsigned*)alloc(64);

  dim3 gtr((HW + 31) / 32, CIN / 32);
  k_tr_feat<<<gtr, 1024, 0, stream>>>(features, ftr);
  k_cvt_w1<<<(9 * COUT * CIN + 255) / 256, 256, 0, stream>>>(W1, w1h);
  k_zeropad<<<1, 16, 0, stream>>>(zpad);
  dim3 gconv((HW + 127) / 128, COUT / 128);
  k_conv<<<gconv, 256, 0, stream>>>(ftr, w1h, b1, (const _Float16*)zpad, ct);
  k_props<<<HW, 64, 0, stream>>>(ct, Ws, bs, Wb, bb, im_info, scores, boxes);
  k_select<<<1, 1024, 0, stream>>>(scores, scal);
  k_compact<<<256, 256, 0, stream>>>(scores, scal, pairs, scal + 2);
  k_sortgather<<<1, 1024, 0, stream>>>(pairs, scal, boxes, props);
  dim3 gnms(NMSB, NMSB);
  k_nms<<<gnms, 64, 0, stream>>>(props, mask);
  k_final<<<1, 256, 0, stream>>>(props, mask, out);
}
